// CrossAttentionWithBias_1546188226913
// MI455X (gfx1250) — matrix operand order repaired and checked by reading, not compiled, not run on hardware
//
#include <hip/hip_runtime.h>

constexpr int kB  = 2;
constexpr int kNQ = 1024;
constexpr int kNK = 2048;
constexpr int kD  = 1024;
constexpr int kH  = 16;
constexpr int kHD = 64;

typedef __attribute__((ext_vector_type(16))) __bf16 bf16x16;
typedef __attribute__((ext_vector_type(2)))  __bf16 bf16x2;
typedef __attribute__((ext_vector_type(8)))  float  floatx8;
typedef __attribute__((ext_vector_type(4)))  unsigned uint32x4;
typedef __attribute__((ext_vector_type(8)))  unsigned uint32x8;

union FragU {
  bf16x16 v;
  uint4   q[2];
};

__device__ inline unsigned short f2bf(float x) {
  unsigned int u = __float_as_uint(x);
  u += 0x7FFFu + ((u >> 16) & 1u);   // round-to-nearest-even
  return (unsigned short)(u >> 16);
}

// Pack two f32 -> two bf16 in one dword (v_cvt_pk_bf16_f32 when available).
__device__ inline unsigned pack_bf16(float lo, float hi) {
#if defined(__has_builtin)
#if __has_builtin(__builtin_amdgcn_cvt_pk_bf16_f32)
  union { bf16x2 v; unsigned u; } cv;
  cv.v = __builtin_amdgcn_cvt_pk_bf16_f32(lo, hi);
  return cv.u;
#else
  return (unsigned)f2bf(lo) | ((unsigned)f2bf(hi) << 16);
#endif
#else
  return (unsigned)f2bf(lo) | ((unsigned)f2bf(hi) << 16);
#endif
}

__device__ inline floatx8 zero8() {
  floatx8 z;
#pragma unroll
  for (int i = 0; i < 8; ++i) z[i] = 0.0f;
  return z;
}

// Async direct global->LDS 16-byte copy (ASYNCcnt-tracked, bypasses VGPRs).
__device__ inline void async_ld_b128(unsigned lds_off, const void* gptr) {
  asm volatile("global_load_async_to_lds_b128 %0, %1, off"
               :: "v"(lds_off), "v"((unsigned long long)(size_t)gptr)
               : "memory");
}
__device__ inline void wait_async0() {
  asm volatile("s_wait_asynccnt 0" ::: "memory");
}

__device__ inline void wait_tensor0() {
#if defined(__has_builtin)
#if __has_builtin(__builtin_amdgcn_s_wait_tensorcnt)
  __builtin_amdgcn_s_wait_tensorcnt(0);
#else
  asm volatile("s_wait_tensorcnt 0" ::: "memory");
#endif
#else
  asm volatile("s_wait_tensorcnt 0" ::: "memory");
#endif
}

// Tensor Data Mover: DMA a rows x 64 bf16 tile (row stride 64 elems) from
// global into LDS with 16B padding after each 128B row (-> 144B LDS pitch).
// D# per CDNA5 ISA 8.3/8.4. One issue per wave; EXEC ignored.
__device__ inline void tdm_load_tile64(unsigned lds_byte, unsigned long long gaddr,
                                       unsigned rows) {
  uint32x4 g0;
  g0[0] = 1u;                                   // count=1, user mode
  g0[1] = lds_byte;                             // lds_addr
  g0[2] = (unsigned)gaddr;                      // global_addr[31:0]
  g0[3] = ((unsigned)(gaddr >> 32) & 0x01FFFFFFu) | (2u << 30);  // [56:32], type=2
  uint32x8 g1;
  g1[0] = (1u << 16)                            // data_size = 2 bytes
        | (1u << 20)                            // pad_enable
        | (4u << 22)                            // pad_interval: 32 DWORDs (128B)
        | (3u << 25);                           // pad_amount: 4 DWORDs (16B)
  g1[1] = (64u << 16);                          // tensor_dim0 = 64  [79:48]
  g1[2] = (rows << 16);                         // tensor_dim0 hi=0, tensor_dim1 lo
  g1[3] = (rows >> 16) | (64u << 16);           // tensor_dim1 hi, tile_dim0 = 64
  g1[4] = rows & 0xFFFFu;                       // tile_dim1 = rows, tile_dim2 = 0
  g1[5] = 64u;                                  // tensor_dim0_stride = 64 elems
  g1[6] = 0u;
  g1[7] = 0u;
  asm volatile("tensor_load_to_lds %0, %1" :: "s"(g0), "s"(g1) : "memory");
}

// A fragment (16x32 bf16, MxK). lanes 0-15: K {0..7,16..23}; lanes 16-31: +8.
__device__ inline bf16x16 frag_a(const unsigned short* base, int stride, int row,
                                 int k0, int lane) {
  const int koff = (lane >= 16) ? 8 : 0;
  const unsigned short* p = base + row * stride + k0 + koff;
  FragU f;
  f.q[0] = *(const uint4*)(p);
  f.q[1] = *(const uint4*)(p + 16);
  return f.v;
}

// B fragment (32x16 bf16, KxN) from [N][K]-major LDS tile.
__device__ inline bf16x16 frag_b(const unsigned short* base, int stride, int col,
                                 int k0, int lane) {
  const int koff = (lane >= 16) ? 8 : 0;
  const unsigned short* p = base + col * stride + k0 + koff;
  FragU f;
  f.q[0] = *(const uint4*)(p);
  f.q[1] = *(const uint4*)(p + 16);
  return f.v;
}

__device__ inline floatx8 wmma_bf16(bf16x16 a, bf16x16 b, floatx8 c) {
  return __builtin_amdgcn_wmma_f32_16x16x32_bf16(false, a, false, b, (short)0, c,
                                                 false, false);
}

// ---------------------------------------------------------------------------
// Per-batch all-masked flag (safe-mask rule).
__global__ void mask_all_kernel(const unsigned char* __restrict__ mask,
                                int* __restrict__ allm) {
  __shared__ int s;
  for (int b = 0; b < kB; ++b) {
    if (threadIdx.x == 0) s = 1;
    __syncthreads();
    int v = 1;
    for (int i = threadIdx.x; i < kNK; i += blockDim.x)
      v &= (mask[b * kNK + i] ? 1 : 0);
    if (!v) atomicAnd(&s, 0);
    __syncthreads();
    if (threadIdx.x == 0) allm[b] = s;
    __syncthreads();
  }
}

// ---------------------------------------------------------------------------
// P = X[M,1024] @ W[1024,1024] + bias, written bf16 head-split [B,H,Ntok,HD].
__global__ __launch_bounds__(256)
void gemm_proj_kernel(const float* __restrict__ X, const float* __restrict__ W,
                      const float* __restrict__ bias,
                      unsigned short* __restrict__ out, int Ntok) {
  __shared__ __align__(16) unsigned short As[128][40];  // [m][k]
  __shared__ __align__(16) unsigned short Bs[128][40];  // [n][k] (transposed)
  const int tid  = threadIdx.x;
  const int lane = tid & 31;
  const int wave = tid >> 5;
  const int m0 = blockIdx.x * 128;
  const int n0 = blockIdx.y * 128;
  const int wm = (wave >> 1) * 32;
  const int wn = (wave & 1) * 64;

  floatx8 acc[2][4];
#pragma unroll
  for (int mi = 0; mi < 2; ++mi)
#pragma unroll
    for (int ni = 0; ni < 4; ++ni) acc[mi][ni] = zero8();

  for (int kk = 0; kk < kD; kk += 32) {
    __syncthreads();
#pragma unroll
    for (int it = 0; it < 4; ++it) {           // A: 128x32 f32 -> bf16 (packed)
      int i = tid + it * 256;
      int row = i >> 3, c4 = (i & 7) * 4;
      float4 f = *(const float4*)(X + (size_t)(m0 + row) * kD + kk + c4);
      *(unsigned*)&As[row][c4 + 0] = pack_bf16(f.x, f.y);
      *(unsigned*)&As[row][c4 + 2] = pack_bf16(f.z, f.w);
    }
#pragma unroll
    for (int it = 0; it < 4; ++it) {           // B: 32x128 f32 -> [n][k] bf16
      int i = tid + it * 256;
      int k = i >> 5, c4 = (i & 31) * 4;
      float4 f = *(const float4*)(W + (size_t)(kk + k) * kD + n0 + c4);
      Bs[c4 + 0][k] = f2bf(f.x);
      Bs[c4 + 1][k] = f2bf(f.y);
      Bs[c4 + 2][k] = f2bf(f.z);
      Bs[c4 + 3][k] = f2bf(f.w);
    }
    __syncthreads();

    bf16x16 af[2], bfr[4];
#pragma unroll
    for (int mi = 0; mi < 2; ++mi)
      af[mi] = frag_a(&As[0][0], 40, wm + mi * 16 + (lane & 15), 0, lane);
#pragma unroll
    for (int ni = 0; ni < 4; ++ni)
      bfr[ni] = frag_b(&Bs[0][0], 40, wn + ni * 16 + (lane & 15), 0, lane);
#pragma unroll
    for (int mi = 0; mi < 2; ++mi)
#pragma unroll
      for (int ni = 0; ni < 4; ++ni)
        acc[mi][ni] = wmma_bf16(af[mi], bfr[ni], acc[mi][ni]);
  }

  const int rbase = (lane >= 16) ? 8 : 0;
  const int cn = lane & 15;
#pragma unroll
  for (int mi = 0; mi < 2; ++mi)
#pragma unroll
    for (int ni = 0; ni < 4; ++ni)
#pragma unroll
      for (int r = 0; r < 8; ++r) {
        int m = m0 + wm + mi * 16 + rbase + r;
        int n = n0 + wn + ni * 16 + cn;
        float v = acc[mi][ni][r] + bias[n];
        int b = m / Ntok, tok = m - b * Ntok;
        int h = n >> 6, hd = n & 63;
        out[(((size_t)b * kH + h) * Ntok + tok) * kHD + hd] = f2bf(v);
      }
}

// ---------------------------------------------------------------------------
// out = X(bf16)[2048,1024] @ W + bias, f32 output.
__global__ __launch_bounds__(256)
void gemm_out_kernel(const unsigned short* __restrict__ X,
                     const float* __restrict__ W, const float* __restrict__ bias,
                     float* __restrict__ out) {
  __shared__ __align__(16) unsigned short As[128][40];
  __shared__ __align__(16) unsigned short Bs[128][40];
  const int tid = threadIdx.x;
  const int lane = tid & 31;
  const int wave = tid >> 5;
  const int m0 = blockIdx.x * 128;
  const int n0 = blockIdx.y * 128;
  const int wm = (wave >> 1) * 32;
  const int wn = (wave & 1) * 64;

  floatx8 acc[2][4];
#pragma unroll
  for (int mi = 0; mi < 2; ++mi)
#pragma unroll
    for (int ni = 0; ni < 4; ++ni) acc[mi][ni] = zero8();

  for (int kk = 0; kk < kD; kk += 32) {
    __syncthreads();
#pragma unroll
    for (int it = 0; it < 2; ++it) {   // A already bf16: async global->LDS
      int i = tid + it * 256;
      int row = i >> 2, c8 = (i & 3) * 8;
      async_ld_b128((unsigned)(size_t)&As[row][c8],
                    X + (size_t)(m0 + row) * kD + kk + c8);
    }
#pragma unroll
    for (int it = 0; it < 4; ++it) {
      int i = tid + it * 256;
      int k = i >> 5, c4 = (i & 31) * 4;
      float4 f = *(const float4*)(W + (size_t)(kk + k) * kD + n0 + c4);
      Bs[c4 + 0][k] = f2bf(f.x);
      Bs[c4 + 1][k] = f2bf(f.y);
      Bs[c4 + 2][k] = f2bf(f.z);
      Bs[c4 + 3][k] = f2bf(f.w);
    }
    wait_async0();
    __syncthreads();

    bf16x16 af[2], bfr[4];
#pragma unroll
    for (int mi = 0; mi < 2; ++mi)
      af[mi] = frag_a(&As[0][0], 40, wm + mi * 16 + (lane & 15), 0, lane);
#pragma unroll
    for (int ni = 0; ni < 4; ++ni)
      bfr[ni] = frag_b(&Bs[0][0], 40, wn + ni * 16 + (lane & 15), 0, lane);
#pragma unroll
    for (int mi = 0; mi < 2; ++mi)
#pragma unroll
      for (int ni = 0; ni < 4; ++ni)
        acc[mi][ni] = wmma_bf16(af[mi], bfr[ni], acc[mi][ni]);
  }

  const int rbase = (lane >= 16) ? 8 : 0;
  const int cn = lane & 15;
#pragma unroll
  for (int mi = 0; mi < 2; ++mi)
#pragma unroll
    for (int ni = 0; ni < 4; ++ni)
#pragma unroll
      for (int r = 0; r < 8; ++r) {
        int m = m0 + wm + mi * 16 + rbase + r;
        int n = n0 + wn + ni * 16 + cn;
        out[(size_t)m * kD + n] = acc[mi][ni][r] + bias[n];
      }
}

// ---------------------------------------------------------------------------
// Flash attention. grid (NQ/64, H, B), block 128 (4 waves, 16 q-rows each).
// Q/K tiles are DMAed by the Tensor Data Mover (one descriptor per tile).
__global__ __launch_bounds__(128)
void attn_kernel(const unsigned short* __restrict__ Qp,
                 const unsigned short* __restrict__ Kp,
                 const unsigned short* __restrict__ Vp,
                 const float* __restrict__ bias,
                 const unsigned char* __restrict__ mask,
                 const int* __restrict__ allm,
                 unsigned short* __restrict__ AO) {
  __shared__ __align__(16) unsigned short Qs[64][72];      // [q][hd]   (TDM, padded)
  __shared__ __align__(16) unsigned short Ks[64][72];      // [kv][hd]  (TDM, padded)
  __shared__ __align__(16) unsigned short Vt[64][72];      // [hd][kv]
  __shared__ __align__(16) unsigned short Ps[4][16][72];   // per-wave P tile

  const int tid = threadIdx.x, lane = tid & 31, wave = tid >> 5;
  const int q0 = blockIdx.x * 64;
  const int h = blockIdx.y, b = blockIdx.z;
  const size_t qbase   = (((size_t)b * kH + h) * kNQ + q0) * kHD;
  const size_t kvstart = ((size_t)b * kH + h) * kNK * kHD;
  const int halfsel = (lane >= 16) ? 1 : 0;
  const int rbase = halfsel * 8;
  const int cn = lane & 15;
  const int m0w = wave * 16;
  const int allmasked = allm[b];

  if (wave == 0) {                              // Q tile 64x64 via TDM
    tdm_load_tile64((unsigned)(size_t)&Qs[0][0],
                    (unsigned long long)(size_t)(Qp + qbase), 64u);
  }

  float mrow[8], lrow[8];
  floatx8 acc[4];
#pragma unroll
  for (int r = 0; r < 8; ++r) { mrow[r] = -1e30f; lrow[r] = 0.0f; }
#pragma unroll
  for (int ni = 0; ni < 4; ++ni) acc[ni] = zero8();

  for (int kv0 = 0; kv0 < kNK; kv0 += 64) {
    __syncthreads();
    if (wave == 0) {                            // K tile 64x64 via TDM
      tdm_load_tile64((unsigned)(size_t)&Ks[0][0],
                      (unsigned long long)(size_t)(Kp + kvstart +
                                                   (size_t)kv0 * kHD), 64u);
    }
    for (int it = 0; it < 32; ++it) {           // V tile transposed -> [hd][kv]
      int i = tid + it * 128;
      int kv = i >> 6, hd = i & 63;
      Vt[hd][kv] = Vp[kvstart + (size_t)(kv0 + kv) * kHD + hd];
    }
    if (wave == 0) wait_tensor0();              // Q (first iter) + K complete
    __syncthreads();

    // S = Q @ K^T  (HD=64 -> two K=32 steps)
    floatx8 s[4];
#pragma unroll
    for (int ni = 0; ni < 4; ++ni) s[ni] = zero8();
#pragma unroll
    for (int ks = 0; ks < 2; ++ks) {
      bf16x16 aq = frag_a(&Qs[0][0], 72, m0w + cn, ks * 32, lane);
#pragma unroll
      for (int ni = 0; ni < 4; ++ni) {
        bf16x16 bk = frag_b(&Ks[0][0], 72, ni * 16 + cn, ks * 32, lane);
        s[ni] = wmma_bf16(aq, bk, s[ni]);
      }
    }

    // logits = s*scale + bias ; mask ; online softmax
    float p[4][8], tmax[8], tsum[8];
#pragma unroll
    for (int r = 0; r < 8; ++r) tmax[r] = -1e30f;
#pragma unroll
    for (int ni = 0; ni < 4; ++ni) {
      int kg = kv0 + ni * 16 + cn;
      int km = (mask[b * kNK + kg] != 0);
      if (kg == 0 && allmasked) km = 0;
      const float* bp = bias + ((((size_t)b * kH + h) * kNQ + (q0 + m0w + rbase)) *
                                kNK + kg);
#pragma unroll
      for (int r = 0; r < 8; ++r) {
        float v = s[ni][r] * 0.125f + __builtin_nontemporal_load(bp + (size_t)r * kNK);
        if (km) v = -1e30f;
        p[ni][r] = v;
        tmax[r] = fmaxf(tmax[r], v);
      }
    }
#pragma unroll
    for (int d = 1; d < 16; d <<= 1)
#pragma unroll
      for (int r = 0; r < 8; ++r) tmax[r] = fmaxf(tmax[r], __shfl_xor(tmax[r], d, 32));
#pragma unroll
    for (int r = 0; r < 8; ++r) {
      float nm = fmaxf(mrow[r], tmax[r]);
      float sc = __expf(mrow[r] - nm);
      mrow[r] = nm;
      lrow[r] *= sc;
#pragma unroll
      for (int ni = 0; ni < 4; ++ni) acc[ni][r] *= sc;
      tsum[r] = 0.0f;
    }
#pragma unroll
    for (int ni = 0; ni < 4; ++ni)
#pragma unroll
      for (int r = 0; r < 8; ++r) {
        float e = __expf(p[ni][r] - mrow[r]);
        p[ni][r] = e;
        tsum[r] += e;
      }
#pragma unroll
    for (int d = 1; d < 16; d <<= 1)
#pragma unroll
      for (int r = 0; r < 8; ++r) tsum[r] += __shfl_xor(tsum[r], d, 32);
#pragma unroll
    for (int r = 0; r < 8; ++r) lrow[r] += tsum[r];

    // C-layout -> A-layout via per-wave LDS tile
#pragma unroll
    for (int ni = 0; ni < 4; ++ni)
#pragma unroll
      for (int r = 0; r < 8; ++r)
        Ps[wave][rbase + r][ni * 16 + cn] = f2bf(p[ni][r]);
    asm volatile("s_wait_dscnt 0" ::: "memory");

    // O += P @ V  (kv=64 -> two K=32 steps)
#pragma unroll
    for (int ks = 0; ks < 2; ++ks) {
      bf16x16 ap = frag_a(&Ps[wave][0][0], 72, cn, ks * 32, lane);
#pragma unroll
      for (int ni = 0; ni < 4; ++ni) {
        bf16x16 bv = frag_b(&Vt[0][0], 72, ni * 16 + cn, ks * 32, lane);
        acc[ni] = wmma_bf16(ap, bv, acc[ni]);
      }
    }
  }

  // normalize and write bf16 [B, NQ, D] with d = h*64 + hd
#pragma unroll
  for (int r = 0; r < 8; ++r) {
    float inv = 1.0f / lrow[r];
    int qg = q0 + m0w + rbase + r;
#pragma unroll
    for (int ni = 0; ni < 4; ++ni) {
      int dg = h * kHD + ni * 16 + cn;
      AO[((size_t)b * kNQ + qg) * kD + dg] = f2bf(acc[ni][r] * inv);
    }
  }
}

// ---------------------------------------------------------------------------
extern "C" void kernel_launch(void* const* d_in, const int* in_sizes, int n_in,
                              void* d_out, int out_size, void* d_ws, size_t ws_size,
                              hipStream_t stream) {
  (void)in_sizes; (void)n_in; (void)out_size; (void)ws_size;
  const float* q_in   = (const float*)d_in[0];
  const float* k_in   = (const float*)d_in[1];
  const float* v_in   = (const float*)d_in[2];
  const float* abias  = (const float*)d_in[3];
  const unsigned char* kpm = (const unsigned char*)d_in[4];
  const float* Wq = (const float*)d_in[5];
  const float* bq = (const float*)d_in[6];
  const float* Wk = (const float*)d_in[7];
  const float* bk = (const float*)d_in[8];
  const float* Wv = (const float*)d_in[9];
  const float* bv = (const float*)d_in[10];
  const float* Wo = (const float*)d_in[11];
  const float* bo = (const float*)d_in[12];

  char* ws = (char*)d_ws;
  unsigned short* Qp = (unsigned short*)ws; ws += (size_t)kB * kH * kNQ * kHD * 2;
  unsigned short* Kp = (unsigned short*)ws; ws += (size_t)kB * kH * kNK * kHD * 2;
  unsigned short* Vp = (unsigned short*)ws; ws += (size_t)kB * kH * kNK * kHD * 2;
  unsigned short* AO = (unsigned short*)ws; ws += (size_t)kB * kNQ * kD * 2;
  int* allm = (int*)ws;

  mask_all_kernel<<<1, 256, 0, stream>>>(kpm, allm);

  gemm_proj_kernel<<<dim3(kB * kNQ / 128, kD / 128), 256, 0, stream>>>(
      q_in, Wq, bq, Qp, kNQ);
  gemm_proj_kernel<<<dim3(kB * kNK / 128, kD / 128), 256, 0, stream>>>(
      k_in, Wk, bk, Kp, kNK);
  gemm_proj_kernel<<<dim3(kB * kNK / 128, kD / 128), 256, 0, stream>>>(
      v_in, Wv, bv, Vp, kNK);

  attn_kernel<<<dim3(kNQ / 64, kH, kB), 128, 0, stream>>>(
      Qp, Kp, Vp, abias, kpm, allm, AO);

  gemm_out_kernel<<<dim3(kB * kNQ / 128, kD / 128), 256, 0, stream>>>(
      AO, Wo, bo, (float*)d_out);
}
